// LocalAttentionBlock_18356690223118
// MI455X (gfx1250) — compile-verified, matrix operand order checked by reading, not run on hardware
//
#include <hip/hip_runtime.h>

typedef __bf16 bf;
typedef __attribute__((ext_vector_type(8)))  __bf16 v8bf;
typedef __attribute__((ext_vector_type(16))) __bf16 v16bf;
typedef __attribute__((ext_vector_type(8)))  float  v8f;

#define B_  2
#define S_  2048
#define D_  1024
#define H_  16
#define HD_ 64
#define W_  128
#define M_  (B_*S_)   /* 4096 rows */

__device__ __forceinline__ bf f2bf(float f){
  unsigned u = __builtin_bit_cast(unsigned, f);
  u += 0x7fffu + ((u >> 16) & 1u);              // round-to-nearest-even
  unsigned short h = (unsigned short)(u >> 16);
  return __builtin_bit_cast(bf, h);
}

__device__ __forceinline__ v16bf cat16(v8bf lo, v8bf hi){
  return __builtin_shufflevector(lo, hi, 0,1,2,3,4,5,6,7,8,9,10,11,12,13,14,15);
}

__device__ __forceinline__ v8f wmma_bf16(v16bf a, v16bf b, v8f c){
  // D(f32 16x16) = A(16x32 bf16) * B(32x16 bf16) + C
  return __builtin_amdgcn_wmma_f32_16x16x32_bf16(false, a, false, b, (short)0, c, false, false);
}

// ---------------------------------------------------------------- converts
__global__ __launch_bounds__(256) void convert_x_kernel(
    const float* __restrict__ x, bf* __restrict__ xbf, int n)
{
  int i = blockIdx.x * 256 + threadIdx.x;
  if (i < n) xbf[i] = f2bf(x[i]);
}

__global__ __launch_bounds__(256) void convert_w_kernel(
    const float* __restrict__ Wq, const float* __restrict__ Wk,
    const float* __restrict__ Wv, const float* __restrict__ Wo,
    bf* __restrict__ Wt)
{
  const float* Ws[4] = {Wq, Wk, Wv, Wo};
  const int z   = blockIdx.z;
  const int idx = blockIdx.x * 256 + threadIdx.x;   // = n*1024 + k
  const int n = idx >> 10, k = idx & 1023;
  Wt[((size_t)z << 20) + idx] = f2bf(Ws[z][(size_t)k * D_ + n]);  // Wt[n][k] = W[k][n]
}

// ---------------------------------------------------------------- QKV GEMM
// Block = 4 waves, each wave computes a 32x64 strip (same 64 N-columns across the
// block so the B stream stays resident in WGP$). K-loop is software-pipelined with
// double-buffered fragments so loads for step k+32 are in flight during step k's WMMAs.
__global__ __launch_bounds__(128) void gemm_qkv_kernel(
    const bf* __restrict__ X, const bf* __restrict__ Wt0,
    const float* __restrict__ bq, const float* __restrict__ bk, const float* __restrict__ bv,
    bf* __restrict__ Q, bf* __restrict__ Kk, bf* __restrict__ Vt)
{
  const int lane = threadIdx.x & 31;
  const int wid  = threadIdx.x >> 5;
  const int half = lane >> 4;
  const int l15  = lane & 15;
  const int n0 = blockIdx.x * 64;
  const int m0 = blockIdx.y * 128 + wid * 32;
  const int which = blockIdx.z;                 // 0=Q 1=K 2=V
  const bf* Wt = Wt0 + ((size_t)which << 20);
  const float* bias = (which == 0) ? bq : (which == 1) ? bk : bv;
  const int ko = half ? 8 : 0;                  // lane-half K-chunk offset (A/B layout)

  v8f acc[2][4] = {};

  auto loadA = [&](int ks, v16bf a[2]) {
    #pragma unroll
    for (int ms = 0; ms < 2; ++ms){
      const bf* ap = X + (size_t)(m0 + 16*ms + l15) * D_ + ks + ko;
      a[ms] = cat16(*(const v8bf*)ap, *(const v8bf*)(ap + 16));
    }
  };
  auto loadB = [&](int ks, v16bf bfr[4]) {
    #pragma unroll
    for (int j = 0; j < 4; ++j){
      const bf* bp = Wt + (size_t)(n0 + 16*j + l15) * D_ + ks + ko;
      bfr[j] = cat16(*(const v8bf*)bp, *(const v8bf*)(bp + 16));
    }
  };
  auto mma = [&](v16bf a[2], v16bf bfr[4]) {
    #pragma unroll
    for (int ms = 0; ms < 2; ++ms)
      #pragma unroll
      for (int j = 0; j < 4; ++j)
        acc[ms][j] = wmma_bf16(a[ms], bfr[j], acc[ms][j]);
  };

  v16bf aA[2], aB[2], bA[4], bB[4];
  loadA(0, aA); loadB(0, bA);
  #pragma unroll 1
  for (int ks = 0; ks < D_; ks += 64){
    loadA(ks + 32, aB); loadB(ks + 32, bB);     // stage k+1 in flight
    mma(aA, bA);                                 // consume stage k
    if (ks + 64 < D_) { loadA(ks + 64, aA); loadB(ks + 64, bA); }
    mma(aB, bB);
  }

  #pragma unroll
  for (int ms = 0; ms < 2; ++ms)
    #pragma unroll
    for (int j = 0; j < 4; ++j){
      const int n = n0 + 16*j + l15;
      const float bia = bias[n];
      #pragma unroll
      for (int i = 0; i < 8; ++i){
        const int m = m0 + 16*ms + i + (half ? 8 : 0);   // C layout: vgpr i, lane-halves -> M
        bf hv = f2bf(acc[ms][j][i] + bia);
        if (which == 2){                          // V stored transposed per head: (B,H,HD,S)
          const int bb = m / S_, s = m % S_;
          const int hh = n >> 6, d = n & 63;
          Vt[(((size_t)bb * H_ + hh) * HD_ + d) * S_ + s] = hv;
        } else if (which == 0){
          Q[(size_t)m * D_ + n] = hv;
        } else {
          Kk[(size_t)m * D_ + n] = hv;
        }
      }
    }
}

// ---------------------------------------------------------------- attention
// 1 wave per (b, h, 16-query tile); 9 key tiles cover [q0-128, q0+15].
__global__ __launch_bounds__(32) void attn_kernel(
    const bf* __restrict__ Q, const bf* __restrict__ K, const bf* __restrict__ Vt,
    bf* __restrict__ O)
{
  __shared__ __align__(16) bf P[16][160];       // probs staged to re-form A layout (+pad tile)
  const int lane = threadIdx.x;
  const int half = lane >> 4;
  const int l15  = lane & 15;
  const int q0 = blockIdx.x * 16;
  const int h  = blockIdx.y;
  const int b  = blockIdx.z;
  const int col0 = h * HD_;
  const bf* Qb = Q  + (size_t)b * S_ * D_;
  const bf* KB = K  + (size_t)b * S_ * D_;
  const bf* VB = Vt + ((size_t)b * H_ + h) * (size_t)HD_ * S_;
  const int ko = half ? 8 : 0;

  // Q fragment: A layout, two K=32 halves of HD=64; contiguous 16B per lane.
  v16bf aq[2];
  #pragma unroll
  for (int hh = 0; hh < 2; ++hh){
    const bf* p = Qb + (size_t)(q0 + l15) * D_ + col0 + hh*32 + ko;
    aq[hh] = cat16(*(const v8bf*)p, *(const v8bf*)(p + 16));
  }

  // scores: 9 tiles of 16x16, each = 2 WMMAs over HD=64
  v8f sc[9];
  #pragma unroll
  for (int t = 0; t < 9; ++t){
    const int k0 = q0 - 128 + 16*t;
    v8f s = {};
    if (k0 + 15 >= 0){                           // uniform skip of fully-masked tiles
      int krow = k0 + l15; if (krow < 0) krow = 0;
      #pragma unroll
      for (int hh = 0; hh < 2; ++hh){
        const bf* p = KB + (size_t)krow * D_ + col0 + hh*32 + ko;  // B col n = K row
        v16bf bm = cat16(*(const v8bf*)p, *(const v8bf*)(p + 16));
        s = wmma_bf16(aq[hh], bm, s);
      }
    }
    sc[t] = s;
  }

  // mask + scale in registers
  const float scale = 0.125f;                    // 1/sqrt(64)
  #pragma unroll
  for (int t = 0; t < 9; ++t){
    const int key = q0 - 128 + 16*t + l15;
    #pragma unroll
    for (int i = 0; i < 8; ++i){
      const int qq = q0 + i + (half ? 8 : 0);
      const bool valid = (key >= 0) && (key <= qq) && ((qq - key) < W_);
      sc[t][i] = valid ? sc[t][i] * scale : -1.0e9f;
    }
  }

  // softmax per row: row lives in one 16-lane half at fixed vgpr index
  #pragma unroll
  for (int i = 0; i < 8; ++i){
    float mx = sc[0][i];
    #pragma unroll
    for (int t = 1; t < 9; ++t) mx = fmaxf(mx, sc[t][i]);
    #pragma unroll
    for (int off = 8; off >= 1; off >>= 1) mx = fmaxf(mx, __shfl_xor(mx, off, 16));
    float r = 0.0f;
    #pragma unroll
    for (int t = 0; t < 9; ++t){ float e = __expf(sc[t][i] - mx); sc[t][i] = e; r += e; }
    #pragma unroll
    for (int off = 8; off >= 1; off >>= 1) r += __shfl_xor(r, off, 16);
    const float pinv = 1.0f / r;
    #pragma unroll
    for (int t = 0; t < 9; ++t) sc[t][i] *= pinv;
  }

  // stage P (bf16) into LDS; pad 10th tile with zeros (9 tiles -> 5 K=32 pairs)
  for (int idx = lane; idx < 256; idx += 32)
    P[idx >> 4][144 + (idx & 15)] = f2bf(0.0f);
  #pragma unroll
  for (int t = 0; t < 9; ++t)
    #pragma unroll
    for (int i = 0; i < 8; ++i)
      P[i + (half ? 8 : 0)][16*t + l15] = f2bf(sc[t][i]);
  __syncthreads();

  // P @ V : 5 pairs x 4 N-tiles of WMMA
  v8f o[4] = {};
  #pragma unroll
  for (int pp = 0; pp < 5; ++pp){
    const int kk0 = 32 * pp;
    const bf* pa = &P[l15][kk0 + ko];
    v16bf am = cat16(*(const v8bf*)pa, *(const v8bf*)(pa + 16));
    int kg = q0 - 128 + kk0 + ko;
    int c0 = kg;      if (c0 < 0) c0 = 0; if (c0 > S_ - 8) c0 = S_ - 8;   // clamped rows hit P==0
    int c1 = kg + 16; if (c1 < 0) c1 = 0; if (c1 > S_ - 8) c1 = S_ - 8;
    #pragma unroll
    for (int j = 0; j < 4; ++j){
      const bf* vb = VB + (size_t)(16*j + l15) * S_;   // Vt: contiguous along keys
      v16bf bm = cat16(*(const v8bf*)(vb + c0), *(const v8bf*)(vb + c1));
      o[j] = wmma_bf16(am, bm, o[j]);
    }
  }

  #pragma unroll
  for (int j = 0; j < 4; ++j)
    #pragma unroll
    for (int i = 0; i < 8; ++i){
      const int qq = q0 + i + (half ? 8 : 0);
      O[((size_t)b * S_ + qq) * D_ + col0 + 16*j + l15] = f2bf(o[j][i]);
    }
}

// ---------------------------------------------------------------- O-proj GEMM + bias + residual
__global__ __launch_bounds__(128) void gemm_o_kernel(
    const bf* __restrict__ A, const bf* __restrict__ Wt,
    const float* __restrict__ bo, const float* __restrict__ xres,
    float* __restrict__ y)
{
  const int lane = threadIdx.x & 31;
  const int wid  = threadIdx.x >> 5;
  const int half = lane >> 4;
  const int l15  = lane & 15;
  const int n0 = blockIdx.x * 64;
  const int m0 = blockIdx.y * 128 + wid * 32;
  const int ko = half ? 8 : 0;

  v8f acc[2][4] = {};

  auto loadA = [&](int ks, v16bf a[2]) {
    #pragma unroll
    for (int ms = 0; ms < 2; ++ms){
      const bf* ap = A + (size_t)(m0 + 16*ms + l15) * D_ + ks + ko;
      a[ms] = cat16(*(const v8bf*)ap, *(const v8bf*)(ap + 16));
    }
  };
  auto loadB = [&](int ks, v16bf bfr[4]) {
    #pragma unroll
    for (int j = 0; j < 4; ++j){
      const bf* bp = Wt + (size_t)(n0 + 16*j + l15) * D_ + ks + ko;
      bfr[j] = cat16(*(const v8bf*)bp, *(const v8bf*)(bp + 16));
    }
  };
  auto mma = [&](v16bf a[2], v16bf bfr[4]) {
    #pragma unroll
    for (int ms = 0; ms < 2; ++ms)
      #pragma unroll
      for (int j = 0; j < 4; ++j)
        acc[ms][j] = wmma_bf16(a[ms], bfr[j], acc[ms][j]);
  };

  v16bf aA[2], aB[2], bA[4], bB[4];
  loadA(0, aA); loadB(0, bA);
  #pragma unroll 1
  for (int ks = 0; ks < D_; ks += 64){
    loadA(ks + 32, aB); loadB(ks + 32, bB);
    mma(aA, bA);
    if (ks + 64 < D_) { loadA(ks + 64, aA); loadB(ks + 64, bA); }
    mma(aB, bB);
  }

  #pragma unroll
  for (int ms = 0; ms < 2; ++ms)
    #pragma unroll
    for (int j = 0; j < 4; ++j){
      const int n = n0 + 16*j + l15;
      const float bia = bo[n];
      #pragma unroll
      for (int i = 0; i < 8; ++i){
        const int m = m0 + 16*ms + i + (half ? 8 : 0);
        y[(size_t)m * D_ + n] = acc[ms][j][i] + bia + xres[(size_t)m * D_ + n];
      }
    }
}

// ---------------------------------------------------------------- layernorm
__global__ __launch_bounds__(256) void ln_kernel(
    const float* __restrict__ y, const float* __restrict__ gamma,
    const float* __restrict__ beta, float* __restrict__ out)
{
  __shared__ float red[256];
  const int tid = threadIdx.x;
  const int row = blockIdx.x;
  const float4 v = ((const float4*)(y + (size_t)row * D_))[tid];
  red[tid] = v.x + v.y + v.z + v.w;
  __syncthreads();
  for (int off = 128; off > 0; off >>= 1){
    if (tid < off) red[tid] += red[tid + off];
    __syncthreads();
  }
  const float mu = red[0] * (1.0f / D_);
  __syncthreads();
  const float dx0 = v.x - mu, dx1 = v.y - mu, dx2 = v.z - mu, dx3 = v.w - mu;
  red[tid] = dx0*dx0 + dx1*dx1 + dx2*dx2 + dx3*dx3;
  __syncthreads();
  for (int off = 128; off > 0; off >>= 1){
    if (tid < off) red[tid] += red[tid + off];
    __syncthreads();
  }
  const float rs = rsqrtf(red[0] * (1.0f / D_) + 1e-5f);
  const float4 g  = ((const float4*)gamma)[tid];
  const float4 bb = ((const float4*)beta)[tid];
  float4 o;
  o.x = dx0 * rs * g.x + bb.x;
  o.y = dx1 * rs * g.y + bb.y;
  o.z = dx2 * rs * g.z + bb.z;
  o.w = dx3 * rs * g.w + bb.w;
  ((float4*)(out + (size_t)row * D_))[tid] = o;
}

// ---------------------------------------------------------------- launcher
extern "C" void kernel_launch(void* const* d_in, const int* in_sizes, int n_in,
                              void* d_out, int out_size, void* d_ws, size_t ws_size,
                              hipStream_t stream)
{
  (void)in_sizes; (void)n_in; (void)out_size; (void)ws_size;
  const float* x     = (const float*)d_in[0];
  const float* Wq    = (const float*)d_in[1];
  const float* bq    = (const float*)d_in[2];
  const float* Wk    = (const float*)d_in[3];
  const float* bk    = (const float*)d_in[4];
  const float* Wv    = (const float*)d_in[5];
  const float* bv    = (const float*)d_in[6];
  const float* Wo    = (const float*)d_in[7];
  const float* bo    = (const float*)d_in[8];
  const float* gamma = (const float*)d_in[9];
  const float* beta  = (const float*)d_in[10];
  float* out = (float*)d_out;

  char* w = (char*)d_ws;
  bf* xbf = (bf*)w;   w += (size_t)M_ * D_ * 2;        //  8 MB  x in bf16
  bf* Wt  = (bf*)w;   w += (size_t)4 * D_ * D_ * 2;    //  8 MB  4x transposed weights bf16
  bf* Qb  = (bf*)w;   w += (size_t)M_ * D_ * 2;        //  8 MB
  bf* Kb  = (bf*)w;   w += (size_t)M_ * D_ * 2;        //  8 MB
  bf* Vt  = (bf*)w;   w += (size_t)M_ * D_ * 2;        //  8 MB  (B,H,HD,S)
  bf* Ab  = (bf*)w;   w += (size_t)M_ * D_ * 2;        //  8 MB  attention output
  float* yv = (float*)w;                               // 16 MB  pre-LN residual sum

  convert_x_kernel<<<dim3((M_ * D_) / 256), 256, 0, stream>>>(x, xbf, M_ * D_);
  convert_w_kernel<<<dim3((D_ * D_) / 256, 1, 4), 256, 0, stream>>>(Wq, Wk, Wv, Wo, Wt);
  gemm_qkv_kernel<<<dim3(D_ / 64, M_ / 128, 3), 128, 0, stream>>>(xbf, Wt, bq, bk, bv, Qb, Kb, Vt);
  attn_kernel<<<dim3(S_ / 16, H_, B_), 32, 0, stream>>>(Qb, Kb, Vt, Ab);
  gemm_o_kernel<<<dim3(D_ / 64, M_ / 128), 128, 0, stream>>>(Ab, Wt + ((size_t)3 << 20), bo, x, yv);
  ln_kernel<<<dim3(M_), 256, 0, stream>>>(yv, gamma, beta, out);
}
